// Attention_53395033424361
// MI455X (gfx1250) — matrix operand order repaired and checked by reading, not compiled, not run on hardware
//
#include <hip/hip_runtime.h>

// ---------------- problem constants ----------------
#define BQ    4
#define NSEQ  1024
#define DIMC  768
#define NH    8
#define HDIM  96
#define MTOT  (BQ*NSEQ)          // 4096
#define QSCALE 0.1020620726159658f  // 1/sqrt(96)

typedef __attribute__((ext_vector_type(16))) __bf16 v16bf;
typedef __attribute__((ext_vector_type(8)))  __bf16 v8bf;
typedef __attribute__((ext_vector_type(8)))  float  v8f;

__device__ inline unsigned short f2bf(float f) {
  __bf16 h = (__bf16)f;
  return __builtin_bit_cast(unsigned short, h);
}

__device__ inline v8f v8f_zero() {
  v8f z;
#pragma unroll
  for (int i = 0; i < 8; ++i) z[i] = 0.0f;
  return z;
}

__device__ inline v8f wmma_bf16(v16bf a, v16bf b, v8f c) {
  // D = A(16x32 bf16) * B(32x16 bf16) + C(16x16 f32)
  return __builtin_amdgcn_wmma_f32_16x16x32_bf16(
      /*neg_a=*/false, a, /*neg_b=*/false, b,
      /*c_mod=*/(short)0, c, /*reuse_a=*/false, /*reuse_b=*/false);
}

// A-matrix 16x32 bf16 fragment, contiguous order: lane group g = lane>>4 holds
// K = {g*16 .. g*16+15} in elements 0..15 (matches B fragments).
__device__ inline v16bf afrag(const unsigned short* rowbase, int bk) {
  v8bf lo = *(const v8bf*)(rowbase + bk);
  v8bf hi = *(const v8bf*)(rowbase + bk + 8);
  return __builtin_shufflevector(lo, hi, 0,1,2,3,4,5,6,7,8,9,10,11,12,13,14,15);
}

__device__ inline void scatter8(unsigned short* dst, int stride, uint4 v) {
  dst[0*stride] = (unsigned short)(v.x);
  dst[1*stride] = (unsigned short)(v.x >> 16);
  dst[2*stride] = (unsigned short)(v.y);
  dst[3*stride] = (unsigned short)(v.y >> 16);
  dst[4*stride] = (unsigned short)(v.z);
  dst[5*stride] = (unsigned short)(v.z >> 16);
  dst[6*stride] = (unsigned short)(v.w);
  dst[7*stride] = (unsigned short)(v.w >> 16);
}

__device__ inline float redmax16(float x) {
  x = fmaxf(x, __shfl_xor(x, 1, 32));
  x = fmaxf(x, __shfl_xor(x, 2, 32));
  x = fmaxf(x, __shfl_xor(x, 4, 32));
  x = fmaxf(x, __shfl_xor(x, 8, 32));
  return x;
}
__device__ inline float redsum16(float x) {
  x += __shfl_xor(x, 1, 32);
  x += __shfl_xor(x, 2, 32);
  x += __shfl_xor(x, 4, 32);
  x += __shfl_xor(x, 8, 32);
  return x;
}

// -------- build dense bf16 weight from quaternion blocks --------
__global__ void qpack_w(const float* __restrict__ w0, const float* __restrict__ w1,
                        const float* __restrict__ w2, const float* __restrict__ w3,
                        unsigned short* __restrict__ out, int OQ) {
  static const int   srct[4][4] = {{0,1,2,3},{1,0,3,2},{2,3,0,1},{3,2,1,0}};
  static const float sgnt[4][4] = {{ 1.f, 1.f, 1.f, 1.f},
                                   {-1.f, 1.f,-1.f, 1.f},
                                   {-1.f, 1.f, 1.f,-1.f},
                                   {-1.f,-1.f, 1.f, 1.f}};
  int idx = blockIdx.x * 256 + threadIdx.x;
  int NN = 4 * OQ;
  int total = 768 * NN;
  if (idx >= total) return;
  int in = idx / NN, o = idx % NN;
  int ic = in / 192, ir = in % 192;
  int oc = o / OQ, orr = o % OQ;
  const float* ws[4] = {w0, w1, w2, w3};
  float val = ws[srct[ic][oc]][(size_t)ir * OQ + orr] * sgnt[ic][oc];
  out[idx] = f2bf(val);
}

__global__ void qcvt_x(const float* __restrict__ x, unsigned short* __restrict__ out, int n) {
  int i = blockIdx.x * 256 + threadIdx.x;
  if (i < n) out[i] = f2bf(x[i]);
}

// ================= software-pipelined 128x128x32 bf16 GEMM core =================
// 256 threads = 8 waves; wave w: rows (w&3)*16 (+0 and +64), cols (w>>2)*64.
// Global loads for tile k+1 issue before the WMMA burst on tile k (latency hiding).
#define GEMM_CORE(Nn)                                                              \
  __shared__ __attribute__((aligned(32))) unsigned short Al[128][32];              \
  __shared__ __attribute__((aligned(32))) unsigned short Bt[128][32];              \
  const int tid = threadIdx.x;                                                     \
  const int w = tid >> 5, l = tid & 31;                                            \
  const int rg = w & 3, ch = w >> 2;                                               \
  const int g = l >> 4, ln = l & 15;                                               \
  const int m0 = blockIdx.y * 128, n0 = blockIdx.x * 128;                          \
  const int ar0 = tid >> 2, ac0 = (tid & 3) * 8;   /* A chunk 0 */                 \
  const int ar1 = (tid + 256) >> 2, ac1 = ac0;     /* A chunk 1 */                 \
  const int bk0 = tid >> 4, bn0 = (tid & 15) * 8;  /* B chunk 0 */                 \
  const int bk1 = (tid + 256) >> 4, bn1 = bn0;     /* B chunk 1 */                 \
  v8f acc0[4], acc1[4];                                                            \
  _Pragma("unroll") for (int i = 0; i < 4; ++i) { acc0[i] = v8f_zero(); acc1[i] = v8f_zero(); } \
  uint4 aR0 = *(const uint4*)&A[(size_t)(m0 + ar0) * 768 + ac0];                   \
  uint4 aR1 = *(const uint4*)&A[(size_t)(m0 + ar1) * 768 + ac1];                   \
  uint4 bR0 = *(const uint4*)&W[(size_t)bk0 * (Nn) + n0 + bn0];                    \
  uint4 bR1 = *(const uint4*)&W[(size_t)bk1 * (Nn) + n0 + bn1];                    \
  for (int k0 = 0; k0 < 768; k0 += 32) {                                           \
    *(uint4*)&Al[ar0][ac0] = aR0;                                                  \
    *(uint4*)&Al[ar1][ac1] = aR1;                                                  \
    scatter8(&Bt[bn0][bk0], 32, bR0);                                              \
    scatter8(&Bt[bn1][bk1], 32, bR1);                                              \
    __syncthreads();                                                               \
    if (k0 + 32 < 768) { /* issue next tile's global loads before compute */       \
      aR0 = *(const uint4*)&A[(size_t)(m0 + ar0) * 768 + k0 + 32 + ac0];           \
      aR1 = *(const uint4*)&A[(size_t)(m0 + ar1) * 768 + k0 + 32 + ac1];           \
      bR0 = *(const uint4*)&W[(size_t)(k0 + 32 + bk0) * (Nn) + n0 + bn0];          \
      bR1 = *(const uint4*)&W[(size_t)(k0 + 32 + bk1) * (Nn) + n0 + bn1];          \
    }                                                                              \
    v16bf a0 = afrag(&Al[rg * 16 + ln][0], g * 16);                                \
    v16bf a1 = afrag(&Al[64 + rg * 16 + ln][0], g * 16);                           \
    v16bf bf[4];                                                                   \
    _Pragma("unroll") for (int cg = 0; cg < 4; ++cg)                               \
      bf[cg] = *(const v16bf*)&Bt[ch * 64 + cg * 16 + ln][g * 16];                 \
    _Pragma("unroll") for (int cg = 0; cg < 4; ++cg)                               \
      acc0[cg] = wmma_bf16(a0, bf[cg], acc0[cg]);                                  \
    _Pragma("unroll") for (int cg = 0; cg < 4; ++cg)                               \
      acc1[cg] = wmma_bf16(a1, bf[cg], acc1[cg]);                                  \
    __syncthreads();                                                               \
  }

// -------- GEMM1: qkv = x_bf16[4096,768] @ Wqkv[768,2304]; scatter q,k,v --------
__global__ __launch_bounds__(256) void qkv_gemm(const unsigned short* __restrict__ A,
                                                const unsigned short* __restrict__ W,
                                                unsigned short* __restrict__ qb,
                                                unsigned short* __restrict__ kbuf,
                                                unsigned short* __restrict__ vbuf) {
  GEMM_CORE(2304)
  // epilogue: col j -> (which,h,d) per torch reshape [B,N,3,H,96]; q scaled
#pragma unroll
  for (int s = 0; s < 2; ++s) {
    v8f* accp = s ? acc1 : acc0;
#pragma unroll
    for (int cg = 0; cg < 4; ++cg) {
      int j = n0 + ch * 64 + cg * 16 + ln;
      int which = j / 768;
      int h = (j % 768) / 96;
      int d = j % 96;
      unsigned short* dst = (which == 0) ? qb : ((which == 1) ? kbuf : vbuf);
      float sc = (which == 0) ? QSCALE : 1.0f;
#pragma unroll
      for (int v = 0; v < 8; ++v) {
        int m = m0 + s * 64 + rg * 16 + v + 8 * g;
        int bb = m >> 10, n = m & 1023;
        dst[((size_t)(bb * NH + h) * NSEQ + n) * HDIM + d] = f2bf(accp[cg][v] * sc);
      }
    }
  }
}

// -------- GEMM2: out = attn_bf[4096,768] @ Wp[768,768] + bias, fp32 out --------
__global__ __launch_bounds__(256) void out_gemm(const unsigned short* __restrict__ A,
                                                const unsigned short* __restrict__ W,
                                                const float* __restrict__ bias,
                                                float* __restrict__ out) {
  GEMM_CORE(768)
#pragma unroll
  for (int s = 0; s < 2; ++s) {
    v8f* accp = s ? acc1 : acc0;
#pragma unroll
    for (int cg = 0; cg < 4; ++cg) {
      int j = n0 + ch * 64 + cg * 16 + ln;
      float bj = bias[j];
#pragma unroll
      for (int v = 0; v < 8; ++v) {
        int m = m0 + s * 64 + rg * 16 + v + 8 * g;
        out[(size_t)m * 768 + j] = accp[cg][v] + bj;
      }
    }
  }
}

// -------- flash quaternion attention --------
// grid (N/128, B*H), 256 threads (8 waves); wave owns 16 query rows.
// Component c score: s_c = q . permsign_c(k) over 96 dims (blocks of 24).
// Output: out = sum_c softmax(s_c) @ permsign_c(v).
__global__ __launch_bounds__(256) void qattn(const unsigned short* __restrict__ qb,
                                             const unsigned short* __restrict__ kb,
                                             const unsigned short* __restrict__ vb,
                                             unsigned short* __restrict__ attn) {
  __shared__ __attribute__((aligned(32))) unsigned short Qs[128][96];   // 24 KB
  __shared__ __attribute__((aligned(32))) unsigned short Ks[64][96];    // 12 KB (permuted K)
  __shared__ __attribute__((aligned(32))) unsigned short Vt[96][64];    // 12 KB (permuted V^T)
  __shared__ __attribute__((aligned(32))) unsigned short Ps[8][16][64]; // 16 KB per-wave P
  static const int perm4[4][4] = {{0,1,2,3},{1,0,3,2},{2,3,0,1},{3,2,1,0}};
  static const unsigned smK[4][4] = {{0u,0x80008000u,0x80008000u,0x80008000u},
                                     {0u,0u,0u,0x80008000u},
                                     {0u,0x80008000u,0u,0u},
                                     {0u,0u,0x80008000u,0u}};
  static const unsigned smV[4][4] = {{0u,0u,0u,0u},
                                     {0x80008000u,0u,0x80008000u,0u},
                                     {0x80008000u,0u,0u,0x80008000u},
                                     {0x80008000u,0x80008000u,0u,0u}};
  const int tid = threadIdx.x;
  const int w = tid >> 5, l = tid & 31;
  const int g = l >> 4, ln = l & 15;
  const int bh = blockIdx.y;
  const int b = bh >> 3, h = bh & 7;
  const int q0 = blockIdx.x * 128;
  const size_t head = (size_t)bh * NSEQ * HDIM;

  // stage Q tile once (q already scaled)
  for (int c = tid; c < 1536; c += 256) {
    int row = c / 12, d0 = (c % 12) * 8;
    *(uint4*)&Qs[row][d0] = *(const uint4*)&qb[head + (size_t)(q0 + row) * HDIM + d0];
  }

  v8f accO[6];
#pragma unroll
  for (int i = 0; i < 6; ++i) accO[i] = v8f_zero();

  for (int comp = 0; comp < 4; ++comp) {
    v8f accC[6];
#pragma unroll
    for (int i = 0; i < 6; ++i) accC[i] = v8f_zero();
    float mrow[8], lrow[8];
#pragma unroll
    for (int v = 0; v < 8; ++v) { mrow[v] = -1e30f; lrow[v] = 0.0f; }

    for (int kt = 0; kt < 16; ++kt) {
      __syncthreads();  // previous iteration consumed Ks/Vt/Ps
      // stage permuted+signed K tile (row-major) and V tile (transposed)
      for (int c = tid; c < 768; c += 256) {
        int key = c / 12, d0 = (c % 12) * 8;
        int ib = d0 / 24, t0 = d0 % 24;            // 8-chunks never cross 24-blocks
        size_t gsrc = head + (size_t)(kt * 64 + key) * HDIM + perm4[comp][ib] * 24 + t0;
        uint4 kv = *(const uint4*)&kb[gsrc];
        unsigned mk = smK[comp][ib];
        kv.x ^= mk; kv.y ^= mk; kv.z ^= mk; kv.w ^= mk;
        *(uint4*)&Ks[key][d0] = kv;
        uint4 vv = *(const uint4*)&vb[gsrc];
        unsigned mv = smV[comp][ib];
        vv.x ^= mv; vv.y ^= mv; vv.z ^= mv; vv.w ^= mv;
        scatter8(&Vt[d0][key], 64, vv);
      }
      __syncthreads();

      // S = Q @ K'^T : K-dim 96 = 3 wmma, 4 key col-tiles (grouped loads -> wmma burst)
      v8f sacc[4];
#pragma unroll
      for (int t = 0; t < 4; ++t) sacc[t] = v8f_zero();
#pragma unroll
      for (int kk = 0; kk < 3; ++kk) {
        v16bf a = afrag(&Qs[w * 16 + ln][0], kk * 32 + g * 16);
        v16bf bfr[4];
#pragma unroll
        for (int t = 0; t < 4; ++t)
          bfr[t] = *(const v16bf*)&Ks[t * 16 + ln][kk * 32 + g * 16];
#pragma unroll
        for (int t = 0; t < 4; ++t)
          sacc[t] = wmma_bf16(a, bfr[t], sacc[t]);
      }

      // online softmax update; write P (bf16) into per-wave LDS tile
#pragma unroll
      for (int v = 0; v < 8; ++v) {
        float tm = fmaxf(fmaxf(sacc[0][v], sacc[1][v]), fmaxf(sacc[2][v], sacc[3][v]));
        tm = redmax16(tm);
        float nm = fmaxf(mrow[v], tm);
        float es = __expf(mrow[v] - nm);
        lrow[v] *= es;
#pragma unroll
        for (int ct = 0; ct < 6; ++ct) accC[ct][v] *= es;
        float rs = 0.0f;
#pragma unroll
        for (int t = 0; t < 4; ++t) {
          float p = __expf(sacc[t][v] - nm);
          rs += p;
          Ps[w][v + 8 * g][t * 16 + ln] = f2bf(p);
        }
        lrow[v] += redsum16(rs);
        mrow[v] = nm;
      }
      __syncthreads();

      // accC += P[16,64] @ V''[64,96] : 2 K-chunks x 6 col-tiles (3-wide bursts)
#pragma unroll
      for (int ko = 0; ko < 2; ++ko) {
        v16bf a = afrag(&Ps[w][ln][0], ko * 32 + g * 16);
#pragma unroll
        for (int cth = 0; cth < 2; ++cth) {
          v16bf bfr[3];
#pragma unroll
          for (int ci = 0; ci < 3; ++ci)
            bfr[ci] = *(const v16bf*)&Vt[(cth * 3 + ci) * 16 + ln][ko * 32 + g * 16];
#pragma unroll
          for (int ci = 0; ci < 3; ++ci)
            accC[cth * 3 + ci] = wmma_bf16(a, bfr[ci], accC[cth * 3 + ci]);
        }
      }
    }

    // finalize component: divide by row sum, add to shared output accumulator
#pragma unroll
    for (int v = 0; v < 8; ++v) {
      float inv = 1.0f / lrow[v];
#pragma unroll
      for (int ct = 0; ct < 6; ++ct) accO[ct][v] += accC[ct][v] * inv;
    }
  }

  // store [B,N,H*96] (pre-projection layout), bf16
#pragma unroll
  for (int ct = 0; ct < 6; ++ct) {
    int d = ct * 16 + ln;
#pragma unroll
    for (int v = 0; v < 8; ++v) {
      int n = q0 + w * 16 + v + 8 * g;
      attn[((size_t)(b * NSEQ + n)) * DIMC + h * HDIM + d] = f2bf(accO[ct][v]);
    }
  }
}

// ---------------- host launcher ----------------
extern "C" void kernel_launch(void* const* d_in, const int* in_sizes, int n_in,
                              void* d_out, int out_size, void* d_ws, size_t ws_size,
                              hipStream_t stream) {
  (void)in_sizes; (void)n_in; (void)out_size; (void)ws_size;
  const float* x      = (const float*)d_in[0];
  const float* wqkv_r = (const float*)d_in[1];
  const float* wqkv_i = (const float*)d_in[2];
  const float* wqkv_j = (const float*)d_in[3];
  const float* wqkv_k = (const float*)d_in[4];
  const float* wp_r   = (const float*)d_in[5];
  const float* wp_i   = (const float*)d_in[6];
  const float* wp_j   = (const float*)d_in[7];
  const float* wp_k   = (const float*)d_in[8];
  const float* bp     = (const float*)d_in[9];

  size_t off = 0;
  char* ws = (char*)d_ws;
  auto take = [&](size_t bytes) -> void* {
    void* r = ws + off;
    off += (bytes + 255) & ~(size_t)255;
    return r;
  };
  unsigned short* XBF  = (unsigned short*)take((size_t)MTOT * DIMC * 2);        // x bf16
  unsigned short* WQKV = (unsigned short*)take((size_t)DIMC * 3 * DIMC * 2);    // [768,2304]
  unsigned short* WP   = (unsigned short*)take((size_t)DIMC * DIMC * 2);        // [768,768]
  unsigned short* QB   = (unsigned short*)take((size_t)BQ * NH * NSEQ * HDIM * 2);
  unsigned short* KB   = (unsigned short*)take((size_t)BQ * NH * NSEQ * HDIM * 2);
  unsigned short* VB   = (unsigned short*)take((size_t)BQ * NH * NSEQ * HDIM * 2);
  unsigned short* AT   = (unsigned short*)take((size_t)MTOT * DIMC * 2);

  const int nx = MTOT * DIMC;
  qcvt_x<<<(nx + 255) / 256, 256, 0, stream>>>(x, XBF, nx);
  qpack_w<<<(768 * 2304 + 255) / 256, 256, 0, stream>>>(wqkv_r, wqkv_i, wqkv_j, wqkv_k, WQKV, 576);
  qpack_w<<<(768 * 768 + 255) / 256, 256, 0, stream>>>(wp_r, wp_i, wp_j, wp_k, WP, 192);

  qkv_gemm<<<dim3(2304 / 128, MTOT / 128), 256, 0, stream>>>(XBF, WQKV, QB, KB, VB);
  qattn<<<dim3(NSEQ / 128, BQ * NH), 256, 0, stream>>>(QB, KB, VB, AT);
  out_gemm<<<dim3(768 / 128, MTOT / 128), 256, 0, stream>>>(AT, WP, bp, (float*)d_out);
}
